// EIT3Block_57535381897829
// MI455X (gfx1250) — compile-verified, matrix operand order checked by reading, not run on hardware
//
#include <hip/hip_runtime.h>
#include <math.h>

typedef __attribute__((ext_vector_type(16))) _Float16 v16h;
typedef __attribute__((ext_vector_type(8)))  _Float16 v8h;
typedef __attribute__((ext_vector_type(8)))  float    v8f;

// ---- problem constants ----
#define DD   1024
#define HH   16
#define DHD  64
#define BB   4
#define LQ   128
#define LP   4096
#define NTOK (LP + LQ)          // 4224
#define MROWS (BB * LQ)         // 512
#define NEGV (-1.0e30f)
#define INV_SQRT_DH 0.125f
#define DECAYF 0.9f
#define THRESHF 1.0e-4f
#define KEEP_RECENT 256
#define COOLDOWN_STEPS 4

// CDNA5 async global->LDS path (guarded: falls back to direct copies)
#if defined(__AMDGCN__) && __has_builtin(__builtin_amdgcn_global_load_async_to_lds_b128) && __has_builtin(__builtin_amdgcn_s_wait_asynccnt)
#define USE_ASYNC 1
#else
#define USE_ASYNC 0
#endif

#if USE_ASYNC
typedef int v4i_ __attribute__((ext_vector_type(4)));
typedef __attribute__((address_space(1))) v4i_* as1_v4i;
typedef __attribute__((address_space(3))) v4i_* as3_v4i;
#endif

__device__ __forceinline__ void wait_async_le(int tag) {
#if USE_ASYNC
    if (tag == 0) __builtin_amdgcn_s_wait_asynccnt(0);
    else          __builtin_amdgcn_s_wait_asynccnt(4);
#endif
}

// Stage a 16x64 fp16 tile (2048B) from global (row stride DD halfs) into LDS
// (row stride 64 halfs). 32 lanes x 16B x 4 issues.
__device__ __forceinline__ void load_tile_16x64(_Float16* ldst,
                                                const _Float16* gsrc,
                                                int lane) {
    const int r   = lane >> 1;
    const int off = (lane & 1) * 16;   // bytes
    const char* g = (const char*)(gsrc + (size_t)r * DD) + off;
    char*       l = (char*)ldst + r * 128 + off;
#if USE_ASYNC
#pragma unroll
    for (int i = 0; i < 4; ++i) {
        __builtin_amdgcn_global_load_async_to_lds_b128(
            (as1_v4i)(uintptr_t)(g + 32 * i),
            (as3_v4i)(unsigned)(uintptr_t)(l + 32 * i),
            0, 0);
    }
#else
#pragma unroll
    for (int i = 0; i < 4; ++i)
        *(v8h*)(l + 32 * i) = *(const v8h*)(g + 32 * i);
#endif
}

// Build a v16h WMMA fragment from p[0..7] and p[16..23]
__device__ __forceinline__ v16h frag_from(const _Float16* p) {
    v8h lo = *(const v8h*)p;
    v8h hi = *(const v8h*)(p + 16);
    return __builtin_shufflevector(lo, hi, 0, 1, 2, 3, 4, 5, 6, 7,
                                   8, 9, 10, 11, 12, 13, 14, 15);
}

// ---- monotonic float<->uint encoding for atomic min/max ----
__device__ __forceinline__ unsigned fenc(float f) {
    unsigned u = __float_as_uint(f);
    return (u & 0x80000000u) ? ~u : (u | 0x80000000u);
}
__device__ __forceinline__ float fdec(unsigned u) {
    unsigned b = (u & 0x80000000u) ? (u & 0x7FFFFFFFu) : ~u;
    return __uint_as_float(b);
}

__device__ __forceinline__ float dqf(float f, float scale, float zp) {
    float q = rintf(f / scale + zp);
    q = fminf(fmaxf(q, 0.0f), 255.0f);
    return (q - zp) * scale;
}

// =====================================================================
// 0) init
// =====================================================================
__global__ void init_kernel(float* imp, unsigned* mm) {
    int i = blockIdx.x * blockDim.x + threadIdx.x;
    if (i < BB * NTOK) imp[i] = 0.0f;
    if (i == 0) { mm[0] = 0xFFFFFFFFu; mm[1] = 0u; }
}

// =====================================================================
// 1) QKV projection (WMMA): q -> ws fp16, k_new/v_new -> d_out f32
// =====================================================================
__global__ __launch_bounds__(32)
void qkv_proj_kernel(const _Float16* __restrict__ x,
                     const float* __restrict__ Wq,
                     const float* __restrict__ Wk,
                     const float* __restrict__ Wv,
                     _Float16* __restrict__ qbuf,
                     float* __restrict__ kout,
                     float* __restrict__ vout) {
    const int mt = blockIdx.x, nt = blockIdx.y, which = blockIdx.z;
    const float* W = (which == 0) ? Wq : ((which == 1) ? Wk : Wv);
    const int lane = threadIdx.x;
    const int lrow = lane & 15;
    const int lsel = lane >> 4;
    const int khalf = lsel * 8;

    v8f acc = {};
    for (int k0 = 0; k0 < DD; k0 += 32) {
        const _Float16* arow = x + (size_t)(mt * 16 + lrow) * DD + k0 + khalf;
        const float*    brow = W + (size_t)(nt * 16 + lrow) * DD + k0 + khalf;
        v16h a = frag_from(arow);
        v16h bf;
#pragma unroll
        for (int i = 0; i < 8; ++i) {
            bf[i]     = (_Float16)brow[i];
            bf[i + 8] = (_Float16)brow[16 + i];
        }
        acc = __builtin_amdgcn_wmma_f32_16x16x32_f16(false, a, false, bf,
                                                     (short)0, acc, false, false);
    }
#pragma unroll
    for (int r = 0; r < 8; ++r) {
        int m = mt * 16 + lsel * 8 + r;
        int n = nt * 16 + lrow;
        _Float16 hv = (_Float16)acc[r];          // emulate .astype(fp16)
        if (which == 0) {
            qbuf[(size_t)m * DD + n] = hv;
        } else {
            int b = m / LQ, qi = m % LQ;
            float* dst = (which == 1) ? kout : vout;
            dst[((size_t)b * NTOK + LP + qi) * DD + n] = (float)hv;
        }
    }
}

// =====================================================================
// 2) copy past K/V (fp16) into d_out float cache regions
// =====================================================================
__global__ void copy_past_kernel(const _Float16* __restrict__ kp,
                                 const _Float16* __restrict__ vp,
                                 float* __restrict__ kout,
                                 float* __restrict__ vout) {
    const size_t total = (size_t)BB * LP * DD;
    const size_t perb  = (size_t)LP * DD;
    for (size_t i = (size_t)blockIdx.x * blockDim.x + threadIdx.x;
         i < total; i += (size_t)gridDim.x * blockDim.x) {
        size_t b = i / perb, rem = i % perb;
        size_t di = b * (size_t)NTOK * DD + rem;
        kout[di] = (float)kp[i];
        vout[di] = (float)vp[i];
    }
}

// =====================================================================
// 3) global min/max
// =====================================================================
__global__ void minmax_kernel(const float* __restrict__ kv, size_t n,
                              unsigned* __restrict__ mm) {
    float lo = 3.4e38f, hi = -3.4e38f;
    for (size_t i = (size_t)blockIdx.x * blockDim.x + threadIdx.x;
         i < n; i += (size_t)gridDim.x * blockDim.x) {
        float v = kv[i];
        lo = fminf(lo, v);
        hi = fmaxf(hi, v);
    }
#pragma unroll
    for (int off = 16; off > 0; off >>= 1) {
        lo = fminf(lo, __shfl_xor(lo, off, 32));
        hi = fmaxf(hi, __shfl_xor(hi, off, 32));
    }
    if ((threadIdx.x & 31) == 0) {
        atomicMin(&mm[0], fenc(lo));
        atomicMax(&mm[1], fenc(hi));
    }
}

__global__ void finalize_quant_kernel(const unsigned* __restrict__ mm,
                                      float* __restrict__ qp) {
    float lo = fdec(mm[0]);
    float hi = fdec(mm[1]);
    float scale = (hi - lo) / 255.0f;
    qp[0] = scale;
    qp[1] = rintf(-lo / scale);
}

// =====================================================================
// 4) prep: stage fp16 K, dequantized fp16 K and V (one pass)
// =====================================================================
__global__ void prep_kernel(const float* __restrict__ k_p,
                            const float* __restrict__ v_p,
                            const float* __restrict__ qp,
                            _Float16* __restrict__ kh,
                            _Float16* __restrict__ kdq,
                            _Float16* __restrict__ vdq) {
    const float scale = qp[0], zp = qp[1];
    const size_t total = (size_t)BB * NTOK * DD;
    for (size_t i = (size_t)blockIdx.x * blockDim.x + threadIdx.x;
         i < total; i += (size_t)gridDim.x * blockDim.x) {
        float kv = k_p[i];
        kh[i]  = (_Float16)kv;
        kdq[i] = (_Float16)dqf(kv, scale, zp);
        vdq[i] = (_Float16)dqf(v_p[i], scale, zp);
    }
}

// =====================================================================
// 5) controller attention: LDS-tiled, double-buffered K stream
// =====================================================================
__global__ __launch_bounds__(32)
void ctrl_attn_kernel(const _Float16* __restrict__ qbuf,
                      const _Float16* __restrict__ kh,
                      float* __restrict__ imp) {
    __shared__ _Float16 kbuf[2][16 * 64];
    const int qt = blockIdx.x, h = blockIdx.y, b = blockIdx.z;
    const int lane = threadIdx.x;
    const int lrow = lane & 15;
    const int lsel = lane >> 4;

    const _Float16* qrow = qbuf + ((size_t)b * LQ + qt * 16 + lrow) * DD + h * DHD + lsel * 8;
    v16h aq0 = frag_from(qrow);
    v16h aq1 = frag_from(qrow + 32);

    const _Float16* kbase = kh + (size_t)b * NTOK * DD + h * DHD;
    const int NT = NTOK / 16;

    float m[8], s[8];
#pragma unroll
    for (int r = 0; r < 8; ++r) { m[r] = -INFINITY; s[r] = 0.0f; }

    // ---- pass 1: online softmax stats ----
    load_tile_16x64(kbuf[0], kbase, lane);
    for (int kt = 0; kt < NT; ++kt) {
        if (kt + 1 < NT) {
            load_tile_16x64(kbuf[(kt + 1) & 1], kbase + (size_t)(kt + 1) * 16 * DD, lane);
            wait_async_le(4);
        } else {
            wait_async_le(0);
        }
        __syncthreads();
        const _Float16* t = kbuf[kt & 1];
        v16h bk0 = frag_from(t + lrow * 64 + lsel * 8);
        v16h bk1 = frag_from(t + lrow * 64 + 32 + lsel * 8);
        v8f c = {};
        c = __builtin_amdgcn_wmma_f32_16x16x32_f16(false, aq0, false, bk0, (short)0, c, false, false);
        c = __builtin_amdgcn_wmma_f32_16x16x32_f16(false, aq1, false, bk1, (short)0, c, false, false);
        const int col = kt * 16 + lrow;
#pragma unroll
        for (int r = 0; r < 8; ++r) {
            int qglob = LP + qt * 16 + lsel * 8 + r;
            float l = c[r] * INV_SQRT_DH;
            if (col > qglob) l = NEGV;
            c[r] = l;
            float red = l;
            red = fmaxf(red, __shfl_xor(red, 1, 32));
            red = fmaxf(red, __shfl_xor(red, 2, 32));
            red = fmaxf(red, __shfl_xor(red, 4, 32));
            red = fmaxf(red, __shfl_xor(red, 8, 32));
            float newm = fmaxf(m[r], red);
            float e = expf(c[r] - newm);
            e += __shfl_xor(e, 1, 32);
            e += __shfl_xor(e, 2, 32);
            e += __shfl_xor(e, 4, 32);
            e += __shfl_xor(e, 8, 32);
            s[r] = s[r] * expf(m[r] - newm) + e;
            m[r] = newm;
        }
        __syncthreads();
    }

    // ---- pass 2: importance accumulation ----
    const float inv = 1.0f / (float)(HH * LQ);
    load_tile_16x64(kbuf[0], kbase, lane);
    for (int kt = 0; kt < NT; ++kt) {
        if (kt + 1 < NT) {
            load_tile_16x64(kbuf[(kt + 1) & 1], kbase + (size_t)(kt + 1) * 16 * DD, lane);
            wait_async_le(4);
        } else {
            wait_async_le(0);
        }
        __syncthreads();
        const _Float16* t = kbuf[kt & 1];
        v16h bk0 = frag_from(t + lrow * 64 + lsel * 8);
        v16h bk1 = frag_from(t + lrow * 64 + 32 + lsel * 8);
        v8f c = {};
        c = __builtin_amdgcn_wmma_f32_16x16x32_f16(false, aq0, false, bk0, (short)0, c, false, false);
        c = __builtin_amdgcn_wmma_f32_16x16x32_f16(false, aq1, false, bk1, (short)0, c, false, false);
        const int col = kt * 16 + lrow;
        float colsum = 0.0f;
#pragma unroll
        for (int r = 0; r < 8; ++r) {
            int qglob = LP + qt * 16 + lsel * 8 + r;
            float l = c[r] * INV_SQRT_DH;
            if (col > qglob) l = NEGV;
            colsum += expf(l - m[r]) / s[r];
        }
        colsum += __shfl_xor(colsum, 16, 32);
        if (lane < 16)
            atomicAdd(&imp[(size_t)b * NTOK + col], colsum * inv);
        __syncthreads();
    }
}

// =====================================================================
// 6) controller state update
// =====================================================================
__global__ void state_update_kernel(const float* __restrict__ ema_in,
                                    const int* __restrict__ mask_in,
                                    const int* __restrict__ cd_in,
                                    const float* __restrict__ imp,
                                    float* __restrict__ mask_out,
                                    float* __restrict__ ema_out,
                                    float* __restrict__ cd_out,
                                    int* __restrict__ maskint) {
    int i = blockIdx.x * blockDim.x + threadIdx.x;
    if (i >= BB * NTOK) return;
    int b = i / NTOK, n = i % NTOK;
    float e0 = (n < LP) ? ema_in[b * LP + n] : 0.0f;
    int   m0 = (n < LP) ? mask_in[b * LP + n] : 0;
    int   c0 = (n < LP) ? cd_in[b * LP + n] : 0;
    float e = DECAYF * e0 + (1.0f - DECAYF) * imp[i];
    int cd = max(c0 - 1, 0);
    bool recent = (n >= NTOK - KEEP_RECENT);
    bool evict = (e < THRESHF) && (cd == 0) && !recent && (m0 == 0);
    int nm = evict ? 1 : m0;
    cd = evict ? COOLDOWN_STEPS : cd;
    ema_out[i]  = e;
    mask_out[i] = (float)nm;
    cd_out[i]   = (float)cd;
    maskint[i]  = nm;
}

// =====================================================================
// 7) masked quantized attention (fp16 staged kdq/vdq, LDS tiled)
// =====================================================================
__global__ __launch_bounds__(32)
void attn_kernel(const _Float16* __restrict__ qbuf,
                 const _Float16* __restrict__ kdq,
                 const _Float16* __restrict__ vdq,
                 const int* __restrict__ maskint,
                 float* __restrict__ attn_out) {
    __shared__ _Float16 kbuf[2][16 * 64];
    __shared__ _Float16 vbuf[2][16 * 64];
    __shared__ _Float16 pbuf[16 * 32];
    const int qt = blockIdx.x, h = blockIdx.y, b = blockIdx.z;
    const int lane = threadIdx.x;
    const int lrow = lane & 15;
    const int lsel = lane >> 4;

    const _Float16* qrow = qbuf + ((size_t)b * LQ + qt * 16 + lrow) * DD + h * DHD + lsel * 8;
    v16h aq0 = frag_from(qrow);
    v16h aq1 = frag_from(qrow + 32);

    const _Float16* kbase = kdq + (size_t)b * NTOK * DD + h * DHD;
    const _Float16* vbase = vdq + (size_t)b * NTOK * DD + h * DHD;
    const int*      mrow  = maskint + (size_t)b * NTOK;
    const int NT = NTOK / 16;

    float m[8], s[8];
#pragma unroll
    for (int r = 0; r < 8; ++r) { m[r] = -INFINITY; s[r] = 0.0f; }

    // ---- pass 1: online softmax stats over dequantized K ----
    load_tile_16x64(kbuf[0], kbase, lane);
    for (int kt = 0; kt < NT; ++kt) {
        if (kt + 1 < NT) {
            load_tile_16x64(kbuf[(kt + 1) & 1], kbase + (size_t)(kt + 1) * 16 * DD, lane);
            wait_async_le(4);
        } else {
            wait_async_le(0);
        }
        __syncthreads();
        const _Float16* t = kbuf[kt & 1];
        v16h bk0 = frag_from(t + lrow * 64 + lsel * 8);
        v16h bk1 = frag_from(t + lrow * 64 + 32 + lsel * 8);
        v8f c = {};
        c = __builtin_amdgcn_wmma_f32_16x16x32_f16(false, aq0, false, bk0, (short)0, c, false, false);
        c = __builtin_amdgcn_wmma_f32_16x16x32_f16(false, aq1, false, bk1, (short)0, c, false, false);
        const int col = kt * 16 + lrow;
        const int mk = mrow[col];
#pragma unroll
        for (int r = 0; r < 8; ++r) {
            int qglob = LP + qt * 16 + lsel * 8 + r;
            float l = c[r] * INV_SQRT_DH;
            if (col > qglob || mk != 0) l = NEGV;
            c[r] = l;
            float red = l;
            red = fmaxf(red, __shfl_xor(red, 1, 32));
            red = fmaxf(red, __shfl_xor(red, 2, 32));
            red = fmaxf(red, __shfl_xor(red, 4, 32));
            red = fmaxf(red, __shfl_xor(red, 8, 32));
            float newm = fmaxf(m[r], red);
            float e = expf(c[r] - newm);
            e += __shfl_xor(e, 1, 32);
            e += __shfl_xor(e, 2, 32);
            e += __shfl_xor(e, 4, 32);
            e += __shfl_xor(e, 8, 32);
            s[r] = s[r] * expf(m[r] - newm) + e;
            m[r] = newm;
        }
        __syncthreads();
    }

    // ---- pass 2: P @ Vdq over 32-key chunks ----
    v8f o0 = {}, o1 = {}, o2 = {}, o3 = {};
    for (int kc = 0; kc < NTOK; kc += 32) {
        load_tile_16x64(kbuf[0], kbase + (size_t)kc * DD, lane);
        load_tile_16x64(kbuf[1], kbase + (size_t)(kc + 16) * DD, lane);
        load_tile_16x64(vbuf[0], vbase + (size_t)kc * DD, lane);
        load_tile_16x64(vbuf[1], vbase + (size_t)(kc + 16) * DD, lane);
        wait_async_le(0);
        __syncthreads();
#pragma unroll
        for (int half = 0; half < 2; ++half) {
            const _Float16* t = kbuf[half];
            v16h bk0 = frag_from(t + lrow * 64 + lsel * 8);
            v16h bk1 = frag_from(t + lrow * 64 + 32 + lsel * 8);
            v8f c = {};
            c = __builtin_amdgcn_wmma_f32_16x16x32_f16(false, aq0, false, bk0, (short)0, c, false, false);
            c = __builtin_amdgcn_wmma_f32_16x16x32_f16(false, aq1, false, bk1, (short)0, c, false, false);
            const int col = kc + half * 16 + lrow;
            const int mk = mrow[col];
#pragma unroll
            for (int r = 0; r < 8; ++r) {
                int qglob = LP + qt * 16 + lsel * 8 + r;
                float l = c[r] * INV_SQRT_DH;
                if (col > qglob || mk != 0) l = NEGV;
                float p = expf(l - m[r]) / s[r];
                pbuf[(lsel * 8 + r) * 32 + half * 16 + lrow] = (_Float16)p;
            }
        }
        __syncthreads();
        v16h ap = frag_from(pbuf + lrow * 32 + lsel * 8);
        __syncthreads();
        v8f* oc[4] = { &o0, &o1, &o2, &o3 };
#pragma unroll
        for (int ct = 0; ct < 4; ++ct) {
            v16h bv;
#pragma unroll
            for (int i = 0; i < 8; ++i) {
                int k1 = lsel * 8 + i;
                int n  = ct * 16 + lrow;
                bv[i]     = vbuf[0][k1 * 64 + n];
                bv[i + 8] = vbuf[1][k1 * 64 + n];
            }
            *oc[ct] = __builtin_amdgcn_wmma_f32_16x16x32_f16(false, ap, false, bv,
                                                             (short)0, *oc[ct], false, false);
        }
        __syncthreads();
    }
    v8f oarr[4] = { o0, o1, o2, o3 };
#pragma unroll
    for (int ct = 0; ct < 4; ++ct) {
#pragma unroll
        for (int r = 0; r < 8; ++r) {
            size_t row = (size_t)b * LQ + qt * 16 + lsel * 8 + r;
            attn_out[row * DD + h * DHD + ct * 16 + lrow] = oarr[ct][r];
        }
    }
}

// =====================================================================
// 8) output projection
// =====================================================================
__global__ __launch_bounds__(32)
void out_proj_kernel(const float* __restrict__ attn_out,
                     const float* __restrict__ Wo,
                     float* __restrict__ outbuf) {
    const int mt = blockIdx.x, nt = blockIdx.y;
    const int lane = threadIdx.x;
    const int lrow = lane & 15;
    const int lsel = lane >> 4;
    const int khalf = lsel * 8;

    v8f acc = {};
    for (int k0 = 0; k0 < DD; k0 += 32) {
        const float* arow = attn_out + (size_t)(mt * 16 + lrow) * DD + k0 + khalf;
        const float* brow = Wo + (size_t)(nt * 16 + lrow) * DD + k0 + khalf;
        v16h a, bf;
#pragma unroll
        for (int i = 0; i < 8; ++i) {
            a[i]      = (_Float16)arow[i];
            a[i + 8]  = (_Float16)arow[16 + i];
            bf[i]     = (_Float16)brow[i];
            bf[i + 8] = (_Float16)brow[16 + i];
        }
        acc = __builtin_amdgcn_wmma_f32_16x16x32_f16(false, a, false, bf,
                                                     (short)0, acc, false, false);
    }
#pragma unroll
    for (int r = 0; r < 8; ++r) {
        size_t m = (size_t)mt * 16 + lsel * 8 + r;
        size_t n = (size_t)nt * 16 + lrow;
        outbuf[m * DD + n] = (float)(_Float16)acc[r];
    }
}

// =====================================================================
// launch
// =====================================================================
extern "C" void kernel_launch(void* const* d_in, const int* in_sizes, int n_in,
                              void* d_out, int out_size, void* d_ws, size_t ws_size,
                              hipStream_t stream) {
    const _Float16* x      = (const _Float16*)d_in[0];
    const _Float16* k_past = (const _Float16*)d_in[1];
    const _Float16* v_past = (const _Float16*)d_in[2];
    const float*    Wq     = (const float*)d_in[3];
    const float*    Wk     = (const float*)d_in[4];
    const float*    Wv     = (const float*)d_in[5];
    const float*    Wo     = (const float*)d_in[6];
    const float*    ema_in = (const float*)d_in[7];
    const int*      mask_in = (const int*)d_in[8];
    const int*      cd_in   = (const int*)d_in[9];

    // d_out float layout in return order
    float* out_p  = (float*)d_out;                                  // [B,LQ,D]
    float* k_p    = out_p + (size_t)BB * LQ * DD;                   // [B,NTOK,D]
    float* v_p    = k_p + (size_t)BB * NTOK * DD;                   // [B,NTOK,D]
    float* mask_p = v_p + (size_t)BB * NTOK * DD;                   // [B,NTOK]
    float* ema_p  = mask_p + (size_t)BB * NTOK;
    float* cd_p   = ema_p + (size_t)BB * NTOK;

    // workspace layout (bytes)
    char* ws = (char*)d_ws;
    const size_t SZ_KV16 = (size_t)BB * NTOK * DD * sizeof(_Float16); // 34,603,008
    _Float16* qbuf    = (_Float16*)(ws);                              // 1,048,576
    _Float16* kh      = (_Float16*)(ws + 1048576);
    _Float16* kdq     = (_Float16*)(ws + 1048576 + SZ_KV16);
    _Float16* vdq     = (_Float16*)(ws + 1048576 + 2 * SZ_KV16);
    char*     tail    = ws + 1048576 + 3 * SZ_KV16;
    float*    attnbuf = (float*)(tail);                               // 2,097,152
    float*    imp     = (float*)(tail + 2097152);                     // 67,584
    int*      maskint = (int*)(tail + 2097152 + 67584);               // 67,584
    unsigned* mm      = (unsigned*)(tail + 2097152 + 2 * 67584);      // 8
    float*    qp      = (float*)(tail + 2097152 + 2 * 67584 + 8);     // 8

    init_kernel<<<(BB * NTOK + 255) / 256, 256, 0, stream>>>(imp, mm);
    qkv_proj_kernel<<<dim3(MROWS / 16, DD / 16, 3), 32, 0, stream>>>(
        x, Wq, Wk, Wv, qbuf, k_p, v_p);
    copy_past_kernel<<<4096, 256, 0, stream>>>(k_past, v_past, k_p, v_p);
    minmax_kernel<<<2048, 256, 0, stream>>>(k_p, (size_t)2 * BB * NTOK * DD, mm);
    finalize_quant_kernel<<<1, 1, 0, stream>>>(mm, qp);
    prep_kernel<<<4096, 256, 0, stream>>>(k_p, v_p, qp, kh, kdq, vdq);
    ctrl_attn_kernel<<<dim3(LQ / 16, HH, BB), 32, 0, stream>>>(qbuf, kh, imp);
    state_update_kernel<<<(BB * NTOK + 255) / 256, 256, 0, stream>>>(
        ema_in, mask_in, cd_in, imp, mask_p, ema_p, cd_p, maskint);
    attn_kernel<<<dim3(LQ / 16, HH, BB), 32, 0, stream>>>(
        qbuf, kdq, vdq, maskint, attnbuf);
    out_proj_kernel<<<dim3(MROWS / 16, DD / 16), 32, 0, stream>>>(attnbuf, Wo, out_p);
}
